// GraphAttentionLayer_13443247637243
// MI455X (gfx1250) — compile-verified, matrix operand order checked by reading, not run on hardware
//
#include <hip/hip_runtime.h>
#include <hip/hip_fp16.h>

typedef __attribute__((ext_vector_type(16))) _Float16     v16h;
typedef __attribute__((ext_vector_type(8)))  float        v8f;
typedef __attribute__((ext_vector_type(4)))  unsigned int v4u;

#define N_NODES 8192
#define F_DIM   256
#define GAT_ALPHA 0.2f
#define NEG_SENT -9.0e15f
#define LDS_STRIDE 40   // f16 elements per staged row (32 data + 8 pad = 80B, kills bank conflicts)

union V16 { v16h v; v4u q[2]; };
union H8  { _Float16 h[8]; v4u q; };

__device__ __forceinline__ void wait_async0() {
#if defined(__has_builtin) && __has_builtin(__builtin_amdgcn_s_wait_asynccnt)
    __builtin_amdgcn_s_wait_asynccnt(0);
#else
    asm volatile("s_wait_asynccnt 0x0" ::: "memory");
#endif
}

// ---------------------------------------------------------------------------
// Kernel 0: convert x -> f16, W -> f16 transposed, zero the f32 accumulator.
// ---------------------------------------------------------------------------
__global__ void gat_prep(const float* __restrict__ x, const float* __restrict__ W,
                         _Float16* __restrict__ xh, _Float16* __restrict__ WT,
                         float* __restrict__ hacc) {
    int t = blockIdx.x * blockDim.x + threadIdx.x;
    if (t < N_NODES * F_DIM) {
        xh[t]   = (_Float16)x[t];
        hacc[t] = 0.0f;
    }
    if (t < F_DIM * F_DIM) {
        int k = t >> 8, n = t & 255;          // W is (k, n) row-major
        WT[n * F_DIM + k] = (_Float16)W[t];   // store transposed
    }
}

// Load a 16x32 f16 tile in WMMA A/B layout from a row-major f16 matrix.
// p = row start + column base for this lane; hi = lane>=16.
__device__ __forceinline__ v16h load_f16_tile(const _Float16* __restrict__ p, int hi) {
    V16 t;
    t.q[0] = *(const v4u*)(p + 8 * hi);        // K = 0..7  (+8 for hi lanes)
    t.q[1] = *(const v4u*)(p + 16 + 8 * hi);   // K = 16..23 (+8 for hi lanes)
    return t.v;
}

// ---------------------------------------------------------------------------
// Kernel 1: h = x @ W  (f16 in, f32 accum), stored TRANSPOSED in f16 (hT).
// One wave per 16x16 output tile; 8192 waves.
// ---------------------------------------------------------------------------
__global__ void __launch_bounds__(256) gat_gemm1(const _Float16* __restrict__ xh,
                                                 const _Float16* __restrict__ WT,
                                                 _Float16* __restrict__ hT) {
    int w    = blockIdx.x * 8 + (threadIdx.x >> 5);
    int lane = threadIdx.x & 31;
    int l16  = lane & 15, hi = lane >> 4;
    int i0 = (w >> 4) * 16;        // row tile
    int n0 = (w & 15) * 16;        // col tile
    const _Float16* arow = xh + (size_t)(i0 + l16) * F_DIM;
    const _Float16* brow = WT + (size_t)(n0 + l16) * F_DIM;
    v8f c = {};
    for (int kk = 0; kk < F_DIM; kk += 32) {
        v16h A = load_f16_tile(arow + kk, hi);
        v16h B = load_f16_tile(brow + kk, hi);
        c = __builtin_amdgcn_wmma_f32_16x16x32_f16(false, A, false, B,
                                                   (short)0, c, false, false);
    }
    // C layout: lane holds col n = l16; VGPR r holds row r + 8*hi.
    // Rows are contiguous in hT (n-major) -> one b128 store per lane.
    H8 o;
#pragma unroll
    for (int r = 0; r < 8; ++r) o.h[r] = (_Float16)c[r];
    *(v4u*)(hT + (size_t)(n0 + l16) * N_NODES + i0 + 8 * hi) = o.q;
}

// ---------------------------------------------------------------------------
// Kernel 2: s1 = h @ a1, s2 = h @ a2 (read hT, lane-contiguous in i).
// ---------------------------------------------------------------------------
__global__ void gat_svec(const _Float16* __restrict__ hT, const float* __restrict__ a,
                         float* __restrict__ s1, float* __restrict__ s2) {
    int i = blockIdx.x * blockDim.x + threadIdx.x;
    float acc1 = 0.0f, acc2 = 0.0f;
    for (int n = 0; n < F_DIM; ++n) {
        float v = (float)hT[(size_t)n * N_NODES + i];
        acc1 += v * a[n];
        acc2 += v * a[F_DIM + n];
    }
    s1[i] = acc1;
    s2[i] = acc2;
}

// ---------------------------------------------------------------------------
// Kernel 3: per-row masked softmax stats (online max / sum-exp). adj pass #1.
// ---------------------------------------------------------------------------
__global__ void gat_rowstats(const int* __restrict__ adj, const float* __restrict__ s1,
                             const float* __restrict__ s2, float* __restrict__ mrow,
                             float* __restrict__ invZ) {
    __shared__ float sm[256];
    __shared__ float sz[256];
    int i = blockIdx.x;
    int t = threadIdx.x;
    float s1i = s1[i];
    const int* arow = adj + (size_t)i * N_NODES;
    float m = NEG_SENT, z = 0.0f;
    for (int k = 0; k < N_NODES / 256; ++k) {
        int j = t + k * 256;
        int av = arow[j];
        float e = s1i + s2[j];
        e = fmaxf(e, GAT_ALPHA * e);              // leaky_relu
        if (av > 0) {
            if (e > m) { z = z * __expf(m - e) + 1.0f; m = e; }
            else       { z += __expf(e - m); }
        }
    }
    sm[t] = m; sz[t] = z;
    __syncthreads();
    for (int s = 128; s > 0; s >>= 1) {
        if (t < s) {
            float m2 = sm[t + s], z2 = sz[t + s];
            float M = fmaxf(sm[t], m2);
            sz[t] = sz[t] * __expf(sm[t] - M) + z2 * __expf(m2 - M);
            sm[t] = M;
        }
        __syncthreads();
    }
    if (t == 0) {
        mrow[i] = sm[0];
        invZ[i] = sz[0] > 0.0f ? 1.0f / sz[0] : 0.0f;
    }
}

// ---------------------------------------------------------------------------
// Kernel 4: h' += softmax(scores) @ h, fused flash-style pass. adj pass #2.
// Block = 8 waves covering 128 rows, all sharing one j-range; the 32x256 h
// tile per j-step (16KB) is staged global->LDS ONCE per block with CDNA5
// async-to-LDS (ASYNCcnt), double-buffered so staging overlaps the exp/mask
// VALU work of the P-tile build. P is built directly in WMMA A layout.
// ---------------------------------------------------------------------------
__global__ void __launch_bounds__(256) gat_attn_gemm(
        const int* __restrict__ adj, const _Float16* __restrict__ hT,
        const float* __restrict__ s1, const float* __restrict__ s2,
        const float* __restrict__ mrow, const float* __restrict__ invZ,
        float* __restrict__ hacc) {
    __shared__ __align__(16) _Float16 smem[2][256 * LDS_STRIDE];   // 2 x 20KB

    int wave = threadIdx.x >> 5;
    int lane = threadIdx.x & 31;
    int l16  = lane & 15, hi = lane >> 4;
    int rg = blockIdx.x >> 2;            // row group (128 rows)
    int sp = blockIdx.x & 3;             // j-range split
    int i0 = rg * 128 + wave * 16;       // this wave's 16-row tile
    int jb = sp * (N_NODES / 4), je = jb + (N_NODES / 4);

    int im = i0 + l16;                   // this lane's A-matrix row
    float s1v = s1[im], mv = mrow[im], izv = invZ[im];
    const int* arow = adj + (size_t)im * N_NODES;

    // Each thread stages one hT row (n = threadIdx.x): 32 f16 = 64B = 4xB128.
    const _Float16* hrow = hT + (size_t)threadIdx.x * N_NODES;
    auto stage = [&](int buf, int j) {
        const _Float16* src = hrow + j;
        unsigned lbase = (unsigned)(unsigned long long)(uintptr_t)
                         &smem[buf][threadIdx.x * LDS_STRIDE];
#pragma unroll
        for (int ch = 0; ch < 4; ++ch) {
            unsigned long long ga = (unsigned long long)(uintptr_t)(src + ch * 8);
            unsigned la = lbase + ch * 16;
            asm volatile("global_load_async_to_lds_b128 %0, %1, off"
                         :: "v"(la), "v"(ga) : "memory");
        }
    };

    // Prologue: stage first tile.
    stage(0, jb);
    wait_async0();
    __syncthreads();

    v8f c[16] = {};
    for (int j = jb; j < je; j += 32) {
        int cb = ((j - jb) >> 5) & 1;
        if (j + 32 < je) stage(cb ^ 1, j + 32);   // prefetch next tile (async)

        // Build A = P[i0:i0+16, j:j+32] in WMMA 16-bit A layout.
        V16 A;
#pragma unroll
        for (int r = 0; r < 8; ++r) {
            int kb = 2 * (r & 3) + 8 * hi + 16 * (r >> 2);
            int jj = j + kb;
            int2   av = *(const int2*)(arow + jj);
            float2 sv = *(const float2*)(s2 + jj);
            float e0 = s1v + sv.x; e0 = fmaxf(e0, GAT_ALPHA * e0);
            float e1 = s1v + sv.y; e1 = fmaxf(e1, GAT_ALPHA * e1);
            float p0 = (av.x > 0) ? __expf(e0 - mv) * izv : 0.0f;
            float p1 = (av.y > 0) ? __expf(e1 - mv) * izv : 0.0f;
            A.v[2 * r]     = (_Float16)p0;
            A.v[2 * r + 1] = (_Float16)p1;
        }

        // B tiles from LDS: smem[cb][n][k], n = nt*16+l16, k in WMMA layout.
#pragma unroll
        for (int nt = 0; nt < 16; ++nt) {
            const _Float16* bp = &smem[cb][(nt * 16 + l16) * LDS_STRIDE + 8 * hi];
            V16 B;
            B.q[0] = *(const v4u*)bp;          // K = 0..7  (+8 for hi lanes)
            B.q[1] = *(const v4u*)(bp + 16);   // K = 16..23 (+8 for hi lanes)
            c[nt] = __builtin_amdgcn_wmma_f32_16x16x32_f16(false, A.v, false, B.v,
                                                           (short)0, c[nt], false, false);
        }

        wait_async0();       // next tile landed in LDS
        __syncthreads();     // all waves done reading cb; cb may be overwritten
    }

#pragma unroll
    for (int nt = 0; nt < 16; ++nt) {
#pragma unroll
        for (int r = 0; r < 8; ++r) {
            int row = i0 + r + 8 * hi;
            int col = nt * 16 + l16;
            unsafeAtomicAdd(&hacc[(size_t)row * F_DIM + col], c[nt][r]);
        }
    }
}

// ---------------------------------------------------------------------------
// Kernel 5: out = elu(hacc)
// ---------------------------------------------------------------------------
__global__ void gat_elu(const float* __restrict__ hacc, float* __restrict__ out) {
    int t = blockIdx.x * blockDim.x + threadIdx.x;
    if (t < N_NODES * F_DIM) {
        float v = hacc[t];
        out[t] = v > 0.0f ? v : expm1f(v);
    }
}

extern "C" void kernel_launch(void* const* d_in, const int* in_sizes, int n_in,
                              void* d_out, int out_size, void* d_ws, size_t ws_size,
                              hipStream_t stream) {
    const float* x   = (const float*)d_in[0];
    const int*   adj = (const int*)d_in[1];
    const float* W   = (const float*)d_in[2];
    const float* a   = (const float*)d_in[3];
    float* out = (float*)d_out;

    char* ws = (char*)d_ws;
    _Float16* xh   = (_Float16*)(ws + 0);          // 4 MB
    _Float16* WT   = (_Float16*)(ws + 4194304);    // 128 KB
    _Float16* hT   = (_Float16*)(ws + 4325376);    // 4 MB (transposed h, f16)
    float*    s1   = (float*)(ws + 8519680);       // 32 KB
    float*    s2   = (float*)(ws + 8552448);       // 32 KB
    float*    mrow = (float*)(ws + 8585216);       // 32 KB
    float*    invZ = (float*)(ws + 8617984);       // 32 KB
    float*    hacc = (float*)(ws + 8650752);       // 8 MB f32 accumulator

    gat_prep     <<<N_NODES * F_DIM / 256, 256, 0, stream>>>(x, W, xh, WT, hacc);
    gat_gemm1    <<<(N_NODES / 16) * (F_DIM / 16) / 8, 256, 0, stream>>>(xh, WT, hT);
    gat_svec     <<<N_NODES / 256, 256, 0, stream>>>(hT, a, s1, s2);
    gat_rowstats <<<N_NODES, 256, 0, stream>>>(adj, s1, s2, mrow, invZ);
    gat_attn_gemm<<<(N_NODES / 128) * 4, 256, 0, stream>>>(adj, hT, s1, s2, mrow, invZ, hacc);
    gat_elu      <<<N_NODES * F_DIM / 256, 256, 0, stream>>>(hacc, out);
}
